// GATClassifier_20590073217144
// MI455X (gfx1250) — matrix operand order repaired and checked by reading, not compiled, not run on hardware
//
#include <hip/hip_runtime.h>
#include <stdint.h>

typedef unsigned short u16;
typedef __attribute__((ext_vector_type(16))) __bf16 v16bf;
typedef __attribute__((ext_vector_type(8)))  float  v8f;

#define HC        256
#define NHEAD     8
#define CDIM      32
#define NGRAPH    64
#define NEG_SLOPE 0.2f
#define EPS_SM    1e-16f
#define INT_MIN_K (-2147483647 - 1)

// ---------- helpers ----------
__device__ __forceinline__ u16 f2bf(float f) {           // round-to-nearest-even
  unsigned u = __float_as_uint(f);
  unsigned r = u + 0x7fffu + ((u >> 16) & 1u);
  return (u16)(r >> 16);
}
__device__ __forceinline__ int f2key(float f) {          // order-preserving int key
  int b = __float_as_int(f);
  return b >= 0 ? b : (b ^ 0x7fffffff);
}
__device__ __forceinline__ float key2f(int k) {
  return __int_as_float(k >= 0 ? k : (k ^ 0x7fffffff));
}

// ---------- conversion kernels ----------
__global__ void k_f32_to_bf16(const float* __restrict__ in, u16* __restrict__ out, long n) {
  long i = (long)blockIdx.x * blockDim.x + threadIdx.x;
  long stride = (long)gridDim.x * blockDim.x;
  for (; i < n; i += stride) out[i] = f2bf(in[i]);
}

// W[K,N] fp32 -> Wt[N,K] bf16
__global__ void k_transpose_bf16(const float* __restrict__ W, u16* __restrict__ Wt,
                                 int K, int N) {
  long i = (long)blockIdx.x * blockDim.x + threadIdx.x;
  if (i >= (long)K * N) return;
  int n = (int)(i / K), k = (int)(i % K);
  Wt[(size_t)n * K + k] = f2bf(W[(size_t)k * N + n]);
}

__global__ void k_fill_int(int* p, int v, long n) {
  long i = (long)blockIdx.x * blockDim.x + threadIdx.x;
  if (i < n) p[i] = v;
}

// ---------- WMMA fragment plumbing ----------
struct Frag { uint4 u[2]; };   // 32 bytes == 16 bf16

__device__ __forceinline__ void load_a(Frag& f, const u16* p) {
  // A 16-bit 16x32 layout: lane holds K chunks [c..c+7] and [c+16..c+23]
  f.u[0] = *reinterpret_cast<const uint4*>(p);
  f.u[1] = *reinterpret_cast<const uint4*>(p + 16);
}
__device__ __forceinline__ void load_b(Frag& f, const u16* p) {
  // B from Wt[N,K]: K chunks [c..c+7] and [c+16..c+23], matching A
  f.u[0] = *reinterpret_cast<const uint4*>(p);
  f.u[1] = *reinterpret_cast<const uint4*>(p + 16);
}
__device__ __forceinline__ v8f wmma_bf16(const Frag& a, const Frag& b, v8f c) {
  return __builtin_amdgcn_wmma_f32_16x16x32_bf16(
      false, __builtin_bit_cast(v16bf, a), false, __builtin_bit_cast(v16bf, b),
      (short)0, c, false, false);
}

// ---------- WMMA GEMM: C[M,N] = A[M,K](bf16) * Bt[N,K](bf16)^T ----------
// Register double-buffered K pipeline; requires K % 64 == 0 (K = 1280 or 256).
__global__ __launch_bounds__(256) void k_gemm_bf16_wmma(
    const u16* __restrict__ A, const u16* __restrict__ Bt, float* __restrict__ C,
    int M, int N, int K) {
  const int lane  = threadIdx.x & 31;
  const int wave  = threadIdx.x >> 5;
  const int mbase = blockIdx.x * 64 + (wave & 3) * 16;
  const int nbase = (wave >> 2) * 128;          // N must be 256
  const int l15   = lane & 15;
  const int lhi   = lane >> 4;

  int arow = mbase + l15; if (arow >= M) arow = M - 1;
  const u16* aptr  = A  + (size_t)arow * K + lhi * 8;
  const u16* bptr0 = Bt + (size_t)(nbase + l15) * K + lhi * 8;

  v8f acc[8] = {};
  Frag A0, A1, B0[8], B1[8];
  const int kiter = K >> 5;

  // prologue: fill both buffers (steps 0 and 1)
  load_a(A0, aptr);
#pragma unroll
  for (int t = 0; t < 8; ++t) load_b(B0[t], bptr0 + (size_t)t * 16 * K);
  load_a(A1, aptr + 32);
#pragma unroll
  for (int t = 0; t < 8; ++t) load_b(B1[t], bptr0 + (size_t)t * 16 * K + 32);

  for (int ki = 0; ki + 2 < kiter; ki += 2) {
    // consume buffer 0 (step ki) while buffer 1 (step ki+1) is in flight
#pragma unroll
    for (int t = 0; t < 8; ++t) acc[t] = wmma_bf16(A0, B0[t], acc[t]);
    // refill buffer 0 with step ki+2
    {
      int ko = (ki + 2) * 32;
      load_a(A0, aptr + ko);
#pragma unroll
      for (int t = 0; t < 8; ++t) load_b(B0[t], bptr0 + (size_t)t * 16 * K + ko);
    }
    // consume buffer 1 (step ki+1) while buffer 0 (step ki+2) is in flight
#pragma unroll
    for (int t = 0; t < 8; ++t) acc[t] = wmma_bf16(A1, B1[t], acc[t]);
    // refill buffer 1 with step ki+3
    {
      int ko = (ki + 3) * 32;
      load_a(A1, aptr + ko);
#pragma unroll
      for (int t = 0; t < 8; ++t) load_b(B1[t], bptr0 + (size_t)t * 16 * K + ko);
    }
  }
  // epilogue: last two k-steps
#pragma unroll
  for (int t = 0; t < 8; ++t) acc[t] = wmma_bf16(A0, B0[t], acc[t]);
#pragma unroll
  for (int t = 0; t < 8; ++t) acc[t] = wmma_bf16(A1, B1[t], acc[t]);

  // store: C/D layout -> row = mbase + (lane>=16)*8 + r, col = nbase + t*16 + (lane&15)
#pragma unroll
  for (int t = 0; t < 8; ++t) {
#pragma unroll
    for (int r = 0; r < 8; ++r) {
      int row = mbase + lhi * 8 + r;
      if (row < M) C[(size_t)row * N + nbase + t * 16 + l15] = acc[t][r];
    }
  }
}

// ---------- attention coefficients: a[n,h] = sum_c h[n,h,c]*att[h,c] ----------
__global__ void k_att(const float* __restrict__ h,
                      const float* __restrict__ attS, const float* __restrict__ attD,
                      float* __restrict__ aS, float* __restrict__ aD, int N) {
  long i = (long)blockIdx.x * blockDim.x + threadIdx.x;
  if (i >= (long)N * NHEAD) return;
  int n = (int)(i >> 3), hh = (int)(i & 7);
  const float* hp = h + (size_t)n * HC + hh * CDIM;
  float s = 0.f, d = 0.f;
#pragma unroll
  for (int c = 0; c < CDIM; ++c) {
    float v = hp[c];
    s += v * attS[hh * CDIM + c];
    d += v * attD[hh * CDIM + c];
  }
  aS[i] = s; aD[i] = d;
}

// ---------- edge pass 1: segment max (ordered-int atomicMax) ----------
__global__ void k_edge_max(const long long* __restrict__ srcE,
                           const long long* __restrict__ dstE,
                           int E, int Etot,
                           const float* __restrict__ aS, const float* __restrict__ aD,
                           int* __restrict__ emax) {
  long i = (long)blockIdx.x * blockDim.x + threadIdx.x;
  if (i >= (long)Etot * NHEAD) return;
  int e = (int)(i >> 3), hh = (int)(i & 7);
  int s = (e < E) ? (int)srcE[e] : (e - E);
  int d = (e < E) ? (int)dstE[e] : (e - E);
  float v = aS[s * NHEAD + hh] + aD[d * NHEAD + hh];
  v = v > 0.f ? v : NEG_SLOPE * v;
  atomicMax(&emax[d * NHEAD + hh], f2key(v));
}

// ---------- edge pass 2: exp weights + denom + unnormalized weighted messages ----------
__global__ __launch_bounds__(256) void k_edge_acc(
    const long long* __restrict__ srcE, const long long* __restrict__ dstE,
    int E, int Etot,
    const float* __restrict__ aS, const float* __restrict__ aD,
    const int* __restrict__ emax, const float* __restrict__ hin,
    float* __restrict__ acc, float* __restrict__ den) {
  int t  = threadIdx.x;        // channel 0..255
  int hh = t >> 5;             // head
  for (int e = blockIdx.x; e < Etot; e += gridDim.x) {
    int s = (e < E) ? (int)srcE[e] : (e - E);
    int d = (e < E) ? (int)dstE[e] : (e - E);
    float v = aS[s * NHEAD + hh] + aD[d * NHEAD + hh];
    v = v > 0.f ? v : NEG_SLOPE * v;
    float w = expf(v - key2f(emax[d * NHEAD + hh]));
    atomicAdd(&acc[(size_t)d * HC + t], hin[(size_t)s * HC + t] * w);
    if ((t & 31) == 0) atomicAdd(&den[d * NHEAD + hh], w);
  }
}

// ---------- finalize: out = relu(acc/(den+eps) + bias); optional bf16 copy ----------
__global__ void k_finalize(float* __restrict__ acc, const float* __restrict__ den,
                           const float* __restrict__ bias, u16* __restrict__ hbf,
                           long total) {
  long i = (long)blockIdx.x * blockDim.x + threadIdx.x;
  if (i >= total) return;
  int t = (int)(i & (HC - 1));
  long n = i >> 8;
  float v = acc[i] / (den[n * NHEAD + (t >> 5)] + EPS_SM) + bias[t];
  v = fmaxf(v, 0.f);
  acc[i] = v;
  if (hbf) hbf[i] = f2bf(v);
}

// ---------- pooling ----------
__global__ void k_pool(const float* __restrict__ h, const long long* __restrict__ batch,
                       float* __restrict__ pool, int N) {
  long i = (long)blockIdx.x * blockDim.x + threadIdx.x;
  if (i >= (long)N * HC) return;
  long n = i >> 8; int t = (int)(i & (HC - 1));
  atomicAdd(&pool[(int)batch[n] * HC + t], h[i]);
}
__global__ void k_count(const long long* __restrict__ batch, float* __restrict__ cnt, int N) {
  int n = blockIdx.x * blockDim.x + threadIdx.x;
  if (n < N) atomicAdd(&cnt[(int)batch[n]], 1.0f);
}

// ---------- final MLP: 64 graphs, 256 -> 128 (relu) -> 1 ----------
__global__ __launch_bounds__(128) void k_mlp(const float* __restrict__ pool,
                                             const float* __restrict__ cnt,
                                             const float* __restrict__ w1,
                                             const float* __restrict__ b1,
                                             const float* __restrict__ w2,
                                             const float* __restrict__ b2,
                                             float* __restrict__ out) {
  __shared__ float z[128];
  int g = blockIdx.x, j = threadIdx.x;
  float invc = 1.0f / fmaxf(cnt[g], 1.0f);
  float a = b1[j];
  for (int k = 0; k < HC; ++k) a += pool[g * HC + k] * invc * w1[k * 128 + j];
  z[j] = fmaxf(a, 0.f);
  __syncthreads();
  if (j == 0) {
    float o = b2[0];
    for (int k = 0; k < 128; ++k) o += z[k] * w2[k];
    out[g] = o;
  }
}

// ==================== launcher ====================
extern "C" void kernel_launch(void* const* d_in, const int* in_sizes, int n_in,
                              void* d_out, int out_size, void* d_ws, size_t ws_size,
                              hipStream_t stream) {
  const float*      x     = (const float*)d_in[0];
  const long long*  eidx  = (const long long*)d_in[1];
  const long long*  batch = (const long long*)d_in[2];
  const float* W1 = (const float*)d_in[3];
  const float* as1 = (const float*)d_in[4];
  const float* ad1 = (const float*)d_in[5];
  const float* b1  = (const float*)d_in[6];
  const float* W2 = (const float*)d_in[7];
  const float* as2 = (const float*)d_in[8];
  const float* ad2 = (const float*)d_in[9];
  const float* b2  = (const float*)d_in[10];
  const float* l1w = (const float*)d_in[11];
  const float* l1b = (const float*)d_in[12];
  const float* l2w = (const float*)d_in[13];
  const float* l2b = (const float*)d_in[14];

  const int Fin  = 1280;
  const int N    = in_sizes[0] / Fin;   // 50000
  const int E    = in_sizes[1] / 2;     // 800000
  const int Etot = E + N;               // self-loops appended virtually
  const long long* srcp = eidx;
  const long long* dstp = eidx + E;

  // workspace arena (~238 MB)
  char* wsb = (char*)d_ws;
  size_t off = 0;
  auto take = [&](size_t bytes) -> char* {
    char* p = wsb + off; off += (bytes + 255) & ~(size_t)255; return p;
  };
  u16*   xb   = (u16*)  take((size_t)N * Fin * 2);   // also reused as h1 bf16
  u16*   w1t  = (u16*)  take((size_t)Fin * HC * 2);
  u16*   w2t  = (u16*)  take((size_t)HC * HC * 2);
  float* hpre = (float*)take((size_t)N * HC * 4);
  float* acc  = (float*)take((size_t)N * HC * 4);
  float* aS   = (float*)take((size_t)N * NHEAD * 4);
  float* aD   = (float*)take((size_t)N * NHEAD * 4);
  int*   emax = (int*)  take((size_t)N * NHEAD * 4);
  float* den  = (float*)take((size_t)N * NHEAD * 4);
  float* pool = (float*)take((size_t)NGRAPH * HC * 4);
  float* cnt  = (float*)take((size_t)NGRAPH * 4);
  (void)ws_size; (void)n_in; (void)out_size;

  // --- prepare bf16 operands ---
  k_f32_to_bf16<<<4096, 256, 0, stream>>>(x, xb, (long)N * Fin);
  k_transpose_bf16<<<((long)Fin * HC + 255) / 256, 256, 0, stream>>>(W1, w1t, Fin, HC);
  k_transpose_bf16<<<((long)HC * HC + 255) / 256, 256, 0, stream>>>(W2, w2t, HC, HC);

  auto gat_layer = [&](const u16* Abf, const u16* Wt, int K,
                       const float* attS, const float* attD, const float* bias,
                       u16* outBf) {
    k_gemm_bf16_wmma<<<(N + 63) / 64, 256, 0, stream>>>(Abf, Wt, hpre, N, HC, K);
    k_att<<<((long)N * NHEAD + 255) / 256, 256, 0, stream>>>(hpre, attS, attD, aS, aD, N);
    k_fill_int<<<((long)N * NHEAD + 255) / 256, 256, 0, stream>>>(emax, INT_MIN_K, (long)N * NHEAD);
    (void)hipMemsetAsync(den, 0, (size_t)N * NHEAD * 4, stream);
    (void)hipMemsetAsync(acc, 0, (size_t)N * HC * 4, stream);
    k_edge_max<<<((long)Etot * NHEAD + 255) / 256, 256, 0, stream>>>(
        srcp, dstp, E, Etot, aS, aD, emax);
    k_edge_acc<<<Etot, 256, 0, stream>>>(srcp, dstp, E, Etot, aS, aD, emax, hpre, acc, den);
    k_finalize<<<((long)N * HC + 255) / 256, 256, 0, stream>>>(
        acc, den, bias, outBf, (long)N * HC);
  };

  // layer 1: x(bf16) @ W1 ; writes h1 (fp32 in acc) and h1 bf16 into xb
  gat_layer(xb, w1t, Fin, as1, ad1, b1, xb);
  // layer 2: h1(bf16) @ W2 ; h2 fp32 left in acc
  gat_layer(xb, w2t, HC, as2, ad2, b2, nullptr);

  // --- global mean pool + MLP head ---
  (void)hipMemsetAsync(pool, 0, (size_t)NGRAPH * HC * 4, stream);
  (void)hipMemsetAsync(cnt, 0, (size_t)NGRAPH * 4, stream);
  k_pool<<<((long)N * HC + 255) / 256, 256, 0, stream>>>(acc, batch, pool, N);
  k_count<<<(N + 255) / 256, 256, 0, stream>>>(batch, cnt, N);
  k_mlp<<<NGRAPH, 128, 0, stream>>>(pool, cnt, l1w, l1b, l2w, l2b, (float*)d_out);
}
